// DDCModel_15238543966875
// MI455X (gfx1250) — matrix operand order repaired and checked by reading, not compiled, not run on hardware
//
#include <hip/hip_runtime.h>
#include <hip/hip_bf16.h>

// ---------------------------------------------------------------------------
// Problem:  x_{t+1} = T_base x_t + sum_a act[t,a] * (A_a x_t) + b
//           reward[t] = dot(r, x_{t+1}),  N=4096, L=50, A_NUM=4
// Strategy: bf16 copy of [T_base; A_0..A_3] (160 MiB -> resident in 192 MB L2),
// 5 stacked GEMVs per step via v_wmma_f32_16x16x32_bf16 with x in B-fragment
// column 0; K split 4-ways into deterministic partials; fixed-order combine.
// ---------------------------------------------------------------------------

#define DDC_N      4096
#define DDC_NN     ((size_t)DDC_N * DDC_N)          // 16,777,216
#define DDC_M5     (5 * DDC_N)                      // stacked rows: 20480
#define DDC_KSPLIT 4

typedef __attribute__((ext_vector_type(16))) __bf16 v16bf;
typedef __attribute__((ext_vector_type(8)))  float  v8f;

static __device__ __forceinline__ unsigned short ddc_f2bf(float f) {
    unsigned u = __float_as_uint(f);
    unsigned r = u + 0x7fffu + ((u >> 16) & 1u);   // round-to-nearest-even
    return (unsigned short)(r >> 16);
}
static __device__ __forceinline__ unsigned ddc_pack2(float a, float b) {
    return (unsigned)ddc_f2bf(a) | ((unsigned)ddc_f2bf(b) << 16);
}

// ---------------------------------------------------------------------------
// Kernel 1: fp32 -> bf16 conversion of [T_base ; A_0..A_3] into d_ws.
// 8 elements per thread; 5*N*N = 83,886,080 elements (exactly 40960 x 256 x 8).
// ---------------------------------------------------------------------------
__global__ __launch_bounds__(256) void ddc_convert_bf16(
    const float* __restrict__ Tb, const float* __restrict__ Am,
    unsigned short* __restrict__ W)
{
    size_t i8 = ((size_t)blockIdx.x * blockDim.x + threadIdx.x) * 8;
    if (i8 >= 5 * DDC_NN) return;
    const float* src = (i8 < DDC_NN) ? (Tb + i8) : (Am + (i8 - DDC_NN));
    float4 f0 = ((const float4*)src)[0];
    float4 f1 = ((const float4*)src)[1];
    uint4 o = make_uint4(ddc_pack2(f0.x, f0.y), ddc_pack2(f0.z, f0.w),
                         ddc_pack2(f1.x, f1.y), ddc_pack2(f1.z, f1.w));
    *(uint4*)(W + i8) = o;
}

// ---------------------------------------------------------------------------
// Kernel 2: zero the bf16 state x (reference starts from zeros).
// ---------------------------------------------------------------------------
__global__ __launch_bounds__(256) void ddc_init_x(unsigned short* __restrict__ xbf)
{
    int i = blockIdx.x * blockDim.x + threadIdx.x;
    if (i < DDC_N) xbf[i] = 0;   // bf16 +0.0
}

// ---------------------------------------------------------------------------
// Kernel 3: stacked GEMV via WMMA.
//   One wave per 16-row tile of the stacked (5N x N) matrix, per K-chunk.
//   grid = (160, KSPLIT);  block = 256 (8 waves).
//   A fragment (16x32 bf16): lane<16 holds row, K=+0..15;
//                            lane>=16 holds row, K=+16..31.
//   B fragment (32x16 bf16): column 0 only -> lane 0 carries K=0..15,
//                            lane 16 carries K=16..31; all other lanes stay
//                            zero (initialized ONCE before the loop; the
//                            conditional reload only touches lanes 0/16).
//   D column 0 -> lane 0 has M=0..7 in c[0..7], lane 16 has M=8..15.
// FROM_F32=true reads fp32 weights from d_in directly (small-ws fallback).
// ---------------------------------------------------------------------------
template <bool FROM_F32>
__global__ __launch_bounds__(256) void ddc_matvec_wmma(
    const unsigned short* __restrict__ Wbf,
    const float* __restrict__ Tb,
    const float* __restrict__ Am,
    const unsigned short* __restrict__ xbf,
    float* __restrict__ yp)                 // [KSPLIT][5N] partial results
{
    const int lane = threadIdx.x & 31;
    const int half = lane >> 4;     // 0: lanes 0-15, 1: lanes 16-31
    const int l15  = lane & 15;
    const int wave = blockIdx.x * (blockDim.x >> 5) + (threadIdx.x >> 5);
    const int r0   = wave << 4;     // stacked row base in [0, 5N)
    const int kc   = blockIdx.y;    // K chunk
    const int kbeg = kc << 10;      // 1024 K per chunk

    v8f c;
#pragma unroll
    for (int i = 0; i < 8; ++i) c[i] = 0.0f;

    const int row = r0 + l15;
    const unsigned short* wrow = nullptr;
    const float*          frow = nullptr;
    if constexpr (FROM_F32) {
        const int m    = r0 >> 12;          // which matrix (uniform per wave)
        const int iloc = row & (DDC_N - 1); // row within matrix
        frow = (m == 0) ? (Tb + (size_t)iloc * DDC_N)
                        : (Am + ((size_t)(m - 1) * DDC_N + iloc) * DDC_N);
    } else {
        wrow = Wbf + (size_t)row * DDC_N;
    }

    // x base for this lane's role in the B fragment (meaningful on lanes 0/16)
    const unsigned short* xlane = xbf + kbeg + half * 16;

    // B fragment: zero everywhere except lanes 0/16; zero lanes are
    // loop-invariant, so initialize once.
    union { v16bf v; uint4 u[2]; } B;
    B.u[0] = make_uint4(0u, 0u, 0u, 0u);
    B.u[1] = make_uint4(0u, 0u, 0u, 0u);
    const bool isColLane = (l15 == 0);

    for (int kk = 0; kk < 1024; kk += 32) {
        const int ka = kbeg + kk + half * 16;
        union { v16bf v; uint4 u[2]; } A;
        if constexpr (FROM_F32) {
            float4 f0 = *(const float4*)(frow + ka);
            float4 f1 = *(const float4*)(frow + ka + 4);
            float4 f2 = *(const float4*)(frow + ka + 8);
            float4 f3 = *(const float4*)(frow + ka + 12);
            A.u[0] = make_uint4(ddc_pack2(f0.x, f0.y), ddc_pack2(f0.z, f0.w),
                                ddc_pack2(f1.x, f1.y), ddc_pack2(f1.z, f1.w));
            A.u[1] = make_uint4(ddc_pack2(f2.x, f2.y), ddc_pack2(f2.z, f2.w),
                                ddc_pack2(f3.x, f3.y), ddc_pack2(f3.z, f3.w));
        } else {
            A.u[0] = *(const uint4*)(wrow + ka);
            A.u[1] = *(const uint4*)(wrow + ka + 8);
        }

        if (isColLane) {  // only column N=0 is nonzero (lanes 0 and 16)
            const uint4* xb = (const uint4*)(xlane + kk);
            B.u[0] = xb[0];
            B.u[1] = xb[1];
        }

        c = __builtin_amdgcn_wmma_f32_16x16x32_bf16(
                /*neg_a=*/false, A.v, /*neg_b=*/false, B.v,
                /*c_mod=*/(short)0, c, /*reuse_a=*/false, /*reuse_b=*/false);
    }

    if (isColLane) {
        float* dst = yp + (size_t)kc * DDC_M5 + r0 + half * 8;
#pragma unroll
        for (int i = 0; i < 8; ++i) dst[i] = c[i];
    }
}

// ---------------------------------------------------------------------------
// Kernel 4: per-step combine + reward.
//   x_new[i] = b[i] + sum_kc ( yp[kc][i] + sum_a act[t,a]*yp[kc][(1+a)N + i] )
//   reward[t] = dot(r, x_new);  store x_new as bf16 for the next step.
// Single 1024-thread workgroup, fixed-order LDS tree reduction (deterministic).
// ---------------------------------------------------------------------------
__global__ __launch_bounds__(1024) void ddc_step_update(
    const float* __restrict__ yp,
    const float* __restrict__ bvec,
    const float* __restrict__ rvec,
    const float* __restrict__ traj,   // [L][4]
    int t,
    unsigned short* __restrict__ xbf,
    float* __restrict__ out)
{
    __shared__ float red[1024];
    const int tid = threadIdx.x;
    const float a0 = traj[t * 4 + 0];
    const float a1 = traj[t * 4 + 1];
    const float a2 = traj[t * 4 + 2];
    const float a3 = traj[t * 4 + 3];

    float acc = 0.0f;
#pragma unroll
    for (int j = 0; j < DDC_N / 1024; ++j) {
        const int i = tid + j * 1024;
        float v = bvec[i];
#pragma unroll
        for (int kc = 0; kc < DDC_KSPLIT; ++kc) {
            const float* p = yp + (size_t)kc * DDC_M5;
            v += p[i]
               + a0 * p[1 * DDC_N + i]
               + a1 * p[2 * DDC_N + i]
               + a2 * p[3 * DDC_N + i]
               + a3 * p[4 * DDC_N + i];
        }
        xbf[i] = ddc_f2bf(v);
        acc += rvec[i] * v;
    }

    red[tid] = acc;
    __syncthreads();
#pragma unroll
    for (int s = 512; s > 0; s >>= 1) {
        if (tid < s) red[tid] += red[tid + s];
        __syncthreads();
    }
    if (tid == 0) out[t] = red[0];
}

// ---------------------------------------------------------------------------
// Host launcher
// ---------------------------------------------------------------------------
extern "C" void kernel_launch(void* const* d_in, const int* in_sizes, int n_in,
                              void* d_out, int out_size, void* d_ws, size_t ws_size,
                              hipStream_t stream)
{
    // setup_inputs() order: init_states(unused), trajectories, T_base, A_mats, b, r
    const float* traj = (const float*)d_in[1];
    const float* Tb   = (const float*)d_in[2];
    const float* Am   = (const float*)d_in[3];
    const float* bvec = (const float*)d_in[4];
    const float* rvec = (const float*)d_in[5];
    float* out = (float*)d_out;
    const int L = in_sizes[1] / 4;   // 50

    const size_t WBF_BYTES = 5 * DDC_NN * sizeof(unsigned short); // 160 MiB
    const size_t XBF_BYTES = 8192;                                // bf16 x (padded)
    const size_t YP_BYTES  = (size_t)DDC_KSPLIT * DDC_M5 * sizeof(float);

    char* ws = (char*)d_ws;
    const bool big_ws = (ws_size >= WBF_BYTES + XBF_BYTES + YP_BYTES);

    unsigned short* Wbf;
    unsigned short* xbf;
    float*          yp;
    if (big_ws) {
        Wbf = (unsigned short*)ws;
        xbf = (unsigned short*)(ws + WBF_BYTES);
        yp  = (float*)(ws + WBF_BYTES + XBF_BYTES);
    } else {
        Wbf = nullptr;
        xbf = (unsigned short*)ws;
        yp  = (float*)(ws + XBF_BYTES);
    }

    if (big_ws) {
        // 83,886,080 elems / 8 per thread / 256 per block = 40960 blocks
        ddc_convert_bf16<<<40960, 256, 0, stream>>>(Tb, Am, Wbf);
    }
    ddc_init_x<<<(DDC_N + 255) / 256, 256, 0, stream>>>(xbf);

    const dim3 mvGrid(DDC_M5 / 16 / 8, DDC_KSPLIT); // (160, 4): 1280 row tiles x 4 K chunks
    for (int t = 0; t < L; ++t) {
        if (big_ws) {
            ddc_matvec_wmma<false><<<mvGrid, 256, 0, stream>>>(Wbf, Tb, Am, xbf, yp);
        } else {
            ddc_matvec_wmma<true><<<mvGrid, 256, 0, stream>>>(Wbf, Tb, Am, xbf, yp);
        }
        ddc_step_update<<<1, 1024, 0, stream>>>(yp, bvec, rvec, traj, t, xbf, out);
    }
}
